// GNBlock_41008347742228
// MI455X (gfx1250) — matrix operand order repaired and checked by reading, not compiled, not run on hardware
//
#include <hip/hip_runtime.h>
#include <hip/hip_bf16.h>

typedef __attribute__((ext_vector_type(16))) __bf16 v16bf;
typedef __attribute__((ext_vector_type(8)))  float  v8f;

#define DD    64      // node/edge/message feature dim
#define HMSG  256     // message MLP hidden
#define HUPD  256     // update MLP hidden
#define KMSG  192     // 3*D
#define KUPD  128     // 2*D
#define TILE  128     // rows per workgroup
#define NTHR  256     // 8 wave32s

// Workspace layout (bf16 element offsets for packed weights, then f32 agg):
#define WM1_OFF 0
#define WM2_OFF (KMSG * HMSG)                 // 49152
#define WU1_OFF (WM2_OFF + HMSG * DD)         // 65536
#define WU2_OFF (WU1_OFF + KUPD * HUPD)       // 98304
#define W_TOTAL (WU2_OFF + HUPD * DD)         // 114688 bf16 elems
#define AGG_BYTE_OFF ((size_t)W_TOTAL * 2)    // 229376 bytes (256B aligned)

// ---------------------------------------------------------------------------
// WMMA A fragment (16x32 bf16, row-major LDS source, stride lda), ISA 7.12.2:
//   lane m = lane&15; lanes 0-15 hold K {0..7,16..23}, lanes 16-31 {8..15,24..31}
//   Per lane this is two contiguous 16B chunks -> compiler emits ds_load_b128 x2.
__device__ __forceinline__ v16bf load_a_frag(const __bf16* a, int lda, int lane) {
    const int m   = lane & 15;
    const int khi = (lane >> 4) << 3;
    const __bf16* row = a + m * lda;
    v16bf f;
#pragma unroll
    for (int v = 0; v < 8; ++v) {
        const int k = ((v & 4) << 2) + khi + ((v & 3) << 1);
        f[2 * v]     = row[k];
        f[2 * v + 1] = row[k + 1];
    }
    return f;
}

// B fragments are pre-packed by gn_prep into fragment-native layout:
// fragment (nt,kt) = 512 bf16 = 1024 B; lane's 16 values contiguous (32 B).
// -> single aligned 32-byte vector load (2x global_load_b128).
__device__ __forceinline__ v16bf load_b_frag_packed(const __bf16* base, int frag, int lane) {
    return *(const v16bf*)(base + ((size_t)frag << 9) + (lane << 4));
}

__device__ __forceinline__ v8f wmma_bf16(v16bf a, v16bf b, v8f c) {
    return __builtin_amdgcn_wmma_f32_16x16x32_bf16(
        /*neg_a=*/false, a, /*neg_b=*/false, b,
        /*c_mod=*/(short)0, c, /*reuse_a=*/false, /*reuse_b=*/false);
}

// Branch-free f32 -> bf16 staging of n contiguous elements (n % 4 == 0).
__device__ __forceinline__ void cvt_store(__bf16* dst, const float* src, int n) {
#pragma unroll 4
    for (int i = 0; i < n; i += 4) {
        const float4 f = *(const float4*)(src + i);
        dst[i]     = (__bf16)f.x;
        dst[i + 1] = (__bf16)f.y;
        dst[i + 2] = (__bf16)f.z;
        dst[i + 3] = (__bf16)f.w;
    }
}

// ---------------------------------------------------------------------------
// Kernel 0: zero agg + convert & PACK all weights fp32 -> fragment-native bf16.
// Packed element idx -> (frag, lane, j); frag = nt*KT + kt;
//   n = nt*16 + (lane&15); k = kt*32 + ((j>=8)?16:0) + ((lane>=16)?8:0) + (j&7); src = W[k*N + n].
__global__ void gn_prep(const float* __restrict__ Wm1, const float* __restrict__ Wm2,
                        const float* __restrict__ Wu1, const float* __restrict__ Wu2,
                        __bf16* __restrict__ wbf, float* __restrict__ agg, int aggElems) {
    const int idx = blockIdx.x * blockDim.x + threadIdx.x;
    if (idx < W_TOTAL) {
        const float* src;
        int local, KT, N;
        if (idx < WM2_OFF)      { src = Wm1; local = idx;           KT = 6; N = HMSG; }
        else if (idx < WU1_OFF) { src = Wm2; local = idx - WM2_OFF; KT = 8; N = DD;   }
        else if (idx < WU2_OFF) { src = Wu1; local = idx - WU1_OFF; KT = 4; N = HUPD; }
        else                    { src = Wu2; local = idx - WU2_OFF; KT = 8; N = DD;   }
        const int frag = local >> 9;
        const int rem  = local & 511;
        const int lane = rem >> 4;
        const int j    = rem & 15;
        const int nt   = frag / KT;
        const int kt   = frag - nt * KT;
        const int n    = nt * 16 + (lane & 15);
        const int k    = kt * 32 + ((j & 8) << 1) + ((lane >> 4) << 3) + (j & 7);
        wbf[idx] = (__bf16)src[(size_t)k * N + n];
    }
    if (idx < aggElems) agg[idx] = 0.0f;
}

// ---------------------------------------------------------------------------
// Kernel 1: message MLP over 128-edge tiles + atomic scatter into agg.
__global__ void gn_msg(const float* __restrict__ h, const float* __restrict__ e,
                       const int* __restrict__ senders, const int* __restrict__ receivers,
                       const float* __restrict__ b_m1, const float* __restrict__ b_m2,
                       const __bf16* __restrict__ wbf, float* __restrict__ agg, int nEdges) {
    extern __shared__ char smem[];
    int*    rcv_s   = (int*)smem;                                   // 128 ints
    __bf16* lds_in  = (__bf16*)(smem + 512);                        // 128 x 192
    __bf16* lds_mid = (__bf16*)(smem + 512 + TILE * KMSG * 2);      // 128 x 256

    const int tid = threadIdx.x;
    const int e0  = blockIdx.x * TILE;

    if (tid < TILE) {
        const int edge = e0 + tid;
        rcv_s[tid] = (edge < nEdges) ? receivers[edge] : 0;
    }

    // Stage m_in = [e | h_s | h_r] as bf16: 2 threads per row, segment-split.
    {
        const int r    = tid >> 1;
        const int half = tid & 1;
        const int edge = e0 + r;
        const int ce   = (edge < nEdges) ? edge : 0;
        const int s    = senders[ce];
        const int rv   = receivers[ce];
        __bf16* dst = lds_in + r * KMSG;
        if (edge < nEdges) {
            if (half == 0) {
                cvt_store(dst,      e + (size_t)ce * DD, DD);       // cols 0..63
                cvt_store(dst + DD, h + (size_t)s  * DD, 32);       // cols 64..95
            } else {
                cvt_store(dst + 96,     h + (size_t)s  * DD + 32, 32);  // 96..127
                cvt_store(dst + 2 * DD, h + (size_t)rv * DD, DD);       // 128..191
            }
        } else {
            __bf16* z = dst + half * 96;
            for (int i = 0; i < 96; ++i) z[i] = (__bf16)0.0f;
        }
    }
    __syncthreads();

    const int wave = tid >> 5;
    const int lane = tid & 31;
    const __bf16* Wm1 = wbf + WM1_OFF;
    const __bf16* Wm2 = wbf + WM2_OFF;

    // ---- layer 1: [128x192] @ [192x256] + b_m1, relu -> lds_mid (bf16) ----
    // Wave owns 2 N-tiles; B fragments register-cached across 8 M-tiles.
#pragma unroll
    for (int nn = 0; nn < 2; ++nn) {
        const int nt = wave * 2 + nn;
        const int n0 = nt * 16;
        v16bf bf[6];
#pragma unroll
        for (int kt = 0; kt < 6; ++kt)
            bf[kt] = load_b_frag_packed(Wm1, nt * 6 + kt, lane);
        const float bias = b_m1[n0 + (lane & 15)];
        const int   col  = n0 + (lane & 15);
        for (int mt = 0; mt < 8; ++mt) {
            v8f acc = {bias, bias, bias, bias, bias, bias, bias, bias};
#pragma unroll
            for (int kt = 0; kt < 6; ++kt) {
                v16bf af = load_a_frag(lds_in + (mt * 16) * KMSG + kt * 32, KMSG, lane);
                acc = wmma_bf16(af, bf[kt], acc);
            }
            const int mbase = mt * 16 + ((lane >> 4) << 3);
#pragma unroll
            for (int rr = 0; rr < 8; ++rr) {
                lds_mid[(mbase + rr) * HMSG + col] = (__bf16)fmaxf(acc[rr], 0.0f);
            }
        }
    }
    __syncthreads();

    // ---- layer 2: [128x256] @ [256x64] + b_m2 -> atomic scatter into agg ----
    // Wave owns N-tile (wave&3) and 4 M-tiles; 8 B fragments register-cached.
    {
        const int nt  = wave & 3;
        const int n0  = nt * 16;
        const int mt0 = (wave >> 2) * 4;
        v16bf bf[8];
#pragma unroll
        for (int kt = 0; kt < 8; ++kt)
            bf[kt] = load_b_frag_packed(Wm2, nt * 8 + kt, lane);
        const int   col  = n0 + (lane & 15);
        const float bias = b_m2[col];
        for (int mt = mt0; mt < mt0 + 4; ++mt) {
            v8f acc = {bias, bias, bias, bias, bias, bias, bias, bias};
#pragma unroll
            for (int kt = 0; kt < 8; ++kt) {
                v16bf af = load_a_frag(lds_mid + (mt * 16) * HMSG + kt * 32, HMSG, lane);
                acc = wmma_bf16(af, bf[kt], acc);
            }
            const int mbase = mt * 16 + ((lane >> 4) << 3);
#pragma unroll
            for (int rr = 0; rr < 8; ++rr) {
                const int row = mbase + rr;
                if (e0 + row < nEdges) {
                    const int rv = rcv_s[row];
                    atomicAdd(agg + (size_t)rv * DD + col, acc[rr]);
                }
            }
        }
    }
}

// ---------------------------------------------------------------------------
// Kernel 2: node update MLP + residual over 128-node tiles.
__global__ void gn_upd(const float* __restrict__ h, const float* __restrict__ agg,
                       const float* __restrict__ b_u1, const float* __restrict__ b_u2,
                       const __bf16* __restrict__ wbf, float* __restrict__ out, int nNodes) {
    extern __shared__ char smem[];
    __bf16* lds_in  = (__bf16*)smem;                          // 128 x 128
    __bf16* lds_mid = (__bf16*)(smem + TILE * KUPD * 2);      // 128 x 256

    const int tid = threadIdx.x;
    const int r0  = blockIdx.x * TILE;

    // Stage u_in = [h | agg]: 2 threads per row, 64 cols each; zero-pad tail.
    {
        const int r    = tid >> 1;
        const int half = tid & 1;
        const int node = r0 + r;
        __bf16* dst = lds_in + r * KUPD + half * DD;
        if (node < nNodes) {
            const float* src = half ? (agg + (size_t)node * DD) : (h + (size_t)node * DD);
            cvt_store(dst, src, DD);
        } else {
            for (int i = 0; i < DD; ++i) dst[i] = (__bf16)0.0f;
        }
    }
    __syncthreads();

    const int wave = tid >> 5;
    const int lane = tid & 31;
    const __bf16* Wu1 = wbf + WU1_OFF;
    const __bf16* Wu2 = wbf + WU2_OFF;

    // ---- layer 1: [128x128] @ [128x256] + b_u1, relu -> lds_mid ----
#pragma unroll
    for (int nn = 0; nn < 2; ++nn) {
        const int nt = wave * 2 + nn;
        const int n0 = nt * 16;
        v16bf bf[4];
#pragma unroll
        for (int kt = 0; kt < 4; ++kt)
            bf[kt] = load_b_frag_packed(Wu1, nt * 4 + kt, lane);
        const float bias = b_u1[n0 + (lane & 15)];
        const int   col  = n0 + (lane & 15);
        for (int mt = 0; mt < 8; ++mt) {
            v8f acc = {bias, bias, bias, bias, bias, bias, bias, bias};
#pragma unroll
            for (int kt = 0; kt < 4; ++kt) {
                v16bf af = load_a_frag(lds_in + (mt * 16) * KUPD + kt * 32, KUPD, lane);
                acc = wmma_bf16(af, bf[kt], acc);
            }
            const int mbase = mt * 16 + ((lane >> 4) << 3);
#pragma unroll
            for (int rr = 0; rr < 8; ++rr) {
                lds_mid[(mbase + rr) * HUPD + col] = (__bf16)fmaxf(acc[rr], 0.0f);
            }
        }
    }
    __syncthreads();

    // ---- layer 2: [128x256] @ [256x64] + b_u2 + residual h -> out ----
    {
        const int nt  = wave & 3;
        const int n0  = nt * 16;
        const int mt0 = (wave >> 2) * 4;
        v16bf bf[8];
#pragma unroll
        for (int kt = 0; kt < 8; ++kt)
            bf[kt] = load_b_frag_packed(Wu2, nt * 8 + kt, lane);
        const int   col  = n0 + (lane & 15);
        const float bias = b_u2[col];
        for (int mt = mt0; mt < mt0 + 4; ++mt) {
            v8f acc = {bias, bias, bias, bias, bias, bias, bias, bias};
#pragma unroll
            for (int kt = 0; kt < 8; ++kt) {
                v16bf af = load_a_frag(lds_mid + (mt * 16) * HUPD + kt * 32, HUPD, lane);
                acc = wmma_bf16(af, bf[kt], acc);
            }
            const int mbase = mt * 16 + ((lane >> 4) << 3);
#pragma unroll
            for (int rr = 0; rr < 8; ++rr) {
                const int row = r0 + mbase + rr;
                if (row < nNodes) {
                    out[(size_t)row * DD + col] = h[(size_t)row * DD + col] + acc[rr];
                }
            }
        }
    }
}

// ---------------------------------------------------------------------------
extern "C" void kernel_launch(void* const* d_in, const int* in_sizes, int n_in,
                              void* d_out, int out_size, void* d_ws, size_t ws_size,
                              hipStream_t stream) {
    (void)n_in; (void)out_size; (void)ws_size;
    const float* h    = (const float*)d_in[0];
    const float* e    = (const float*)d_in[1];
    const int* senders   = (const int*)d_in[2];
    const int* receivers = (const int*)d_in[3];
    const float* W_m1 = (const float*)d_in[4];
    const float* b_m1 = (const float*)d_in[5];
    const float* W_m2 = (const float*)d_in[6];
    const float* b_m2 = (const float*)d_in[7];
    const float* W_u1 = (const float*)d_in[8];
    const float* b_u1 = (const float*)d_in[9];
    const float* W_u2 = (const float*)d_in[10];
    const float* b_u2 = (const float*)d_in[11];
    float* out = (float*)d_out;

    const int nNodes = in_sizes[0] / DD;   // 50000
    const int nEdges = in_sizes[2];        // 800000

    __bf16* wbf = (__bf16*)d_ws;
    float*  agg = (float*)((char*)d_ws + AGG_BYTE_OFF);
    const int aggElems = nNodes * DD;

    // 0) zero agg, convert+pack weights to fragment-native bf16
    const int prepWork = (aggElems > W_TOTAL) ? aggElems : W_TOTAL;
    gn_prep<<<(prepWork + NTHR - 1) / NTHR, NTHR, 0, stream>>>(
        W_m1, W_m2, W_u1, W_u2, wbf, agg, aggElems);

    // 1) message MLP + scatter-add (115200 B dynamic LDS -> 2 WGs / 320KB WGP)
    const int msgBlocks = (nEdges + TILE - 1) / TILE;
    const size_t msgShm = 512 + (size_t)TILE * KMSG * 2 + (size_t)TILE * HMSG * 2;
    gn_msg<<<msgBlocks, NTHR, msgShm, stream>>>(
        h, e, senders, receivers, b_m1, b_m2, wbf, agg, nEdges);

    // 2) node update MLP + residual
    const int updBlocks = (nNodes + TILE - 1) / TILE;
    const size_t updShm = (size_t)TILE * KUPD * 2 + (size_t)TILE * HUPD * 2;
    gn_upd<<<updBlocks, NTHR, updShm, stream>>>(
        h, agg, b_u1, b_u2, wbf, out, nNodes);
}
